// he_gat_86526411145570
// MI455X (gfx1250) — matrix operand order repaired and checked by reading, not compiled, not run on hardware
//
#include <hip/hip_runtime.h>

typedef __attribute__((ext_vector_type(16))) _Float16 v16h;
typedef __attribute__((ext_vector_type(8)))  _Float16 v8h;
typedef __attribute__((ext_vector_type(8)))  float    v8f;
typedef __attribute__((ext_vector_type(4)))  unsigned int v4u;
typedef __attribute__((ext_vector_type(8)))  int      v8i;
typedef __attribute__((ext_vector_type(4)))  int      v4i;

constexpr int GB = 64;       // batches
constexpr int GN = 1024;     // nodes
constexpr int GH = 64;       // features
constexpr int ROWS = 32;     // rows per workgroup (two 16-row WMMA tiles)
constexpr int CHUNK = 32;    // q-chunk = WMMA K for f16
constexpr int THREADS = 128; // 4 waves (wave32)
constexpr int PXS = 40;      // padded half-stride for prob rows (16B-aligned, bank spread)
constexpr int XTS = 40;      // padded half-stride for transposed inp cols

// ---------------------------------------------------------------------------
// Tensor Data Mover: 1-D contiguous global->LDS DMA (f32 dwords).
// D# built per CDNA5 ISA ch.8: group0 {count=1, lds_addr, global_addr, type=2},
// group1 {data_size=4B, tensor_dim0=n, tensor_dim1=1, tile_dim0=n, tile_dim1=1,
//         tensor_dim0_stride=n}. Groups 2/3 zero (<=2D tensor).
// This toolchain (clang-23 / therock-10.0 headers) uses the 6-arg builtin.
// ---------------------------------------------------------------------------
__device__ __forceinline__ void tdm_load_1d(unsigned lds_off, const void* gptr,
                                            unsigned ndw) {
  unsigned long long ga = (unsigned long long)(uintptr_t)gptr;
  v4u g0;
  g0[0] = 1u;                                              // count=1 (valid)
  g0[1] = lds_off;                                         // lds_addr (bytes)
  g0[2] = (unsigned)(ga & 0xffffffffu);                    // global_addr[31:0]
  g0[3] = (unsigned)((ga >> 32) & 0x01ffffffu) | (2u << 30); // [56:32] | type=2
  v8i g1;
  g1[0] = (int)(2u << 16);                                 // data_size = 4 bytes
  g1[1] = (int)((ndw & 0xffffu) << 16);                    // tensor_dim0[15:0]
  g1[2] = (int)(((ndw >> 16) & 0xffffu) | (1u << 16));     // dim0[31:16], tensor_dim1=1
  g1[3] = (int)((ndw & 0xffffu) << 16);                    // tile_dim0 = ndw (<=65535)
  g1[4] = 1;                                               // tile_dim1 = 1
  g1[5] = (int)ndw;                                        // tensor_dim0_stride
  g1[6] = 0;
  g1[7] = 0;
  v4i z4 = {0, 0, 0, 0};
  v8i z8 = {0, 0, 0, 0, 0, 0, 0, 0};
  __builtin_amdgcn_tensor_load_to_lds(g0, g1, z4, z4, z8, 0);
}

// ---------------------------------------------------------------------------
// Kernel 1: range-selected additive score halves.
// ---------------------------------------------------------------------------
__global__ __launch_bounds__(THREADS) void gat_scores(
    const float* __restrict__ inp, const float* __restrict__ inp_none,
    const int* __restrict__ l, const float* __restrict__ a_i,
    const float* __restrict__ a_c, float* __restrict__ wh1,
    float* __restrict__ wh2)
{
  int idx = blockIdx.x * THREADS + threadIdx.x;   // < B*N
  int b = idx >> 10;                              // N == 1024
  int n = idx & (GN - 1);
  int lo = l[2 * b], hi = l[2 * b + 1];
  bool ir = (n >= lo) && (n < hi);
  const float* av = ir ? a_i : a_c;
  const float4* r2 = (const float4*)(inp + (size_t)idx * GH);
  const float4* r1 = ir ? (const float4*)(inp_none + (size_t)idx * GH) : r2;
  const float4* c1 = (const float4*)av;
  const float4* c2 = (const float4*)(av + GH);
  float s1 = 0.f, s2 = 0.f;
  #pragma unroll
  for (int i = 0; i < GH / 4; ++i) {
    float4 x1 = r1[i], x2 = r2[i], a1 = c1[i], a2 = c2[i];
    s1 += x1.x * a1.x + x1.y * a1.y + x1.z * a1.z + x1.w * a1.w;
    s2 += x2.x * a2.x + x2.y * a2.y + x2.z * a2.z + x2.w * a2.w;
  }
  wh1[idx] = s1;
  wh2[idx] = s2;
}

// ---------------------------------------------------------------------------
// Kernel 2: per (batch, 32-row tile): TDM-stage A tile -> masked scores ->
// softmax -> attn @ inp via v_wmma_f32_16x16x32_f16.
// ---------------------------------------------------------------------------
__global__ __launch_bounds__(THREADS) void gat_main(
    const float* __restrict__ inp, const float* __restrict__ Adj,
    const float* __restrict__ wh1g, const float* __restrict__ wh2g,
    float* __restrict__ out)
{
  extern __shared__ char smem[];
  float*    e    = (float*)smem;                 // [ROWS][GN]   128 KB (A tile -> scores)
  float*    wh2  = e + ROWS * GN;                // [GN]           4 KB
  float*    wh1  = wh2 + GN;                     // [ROWS]
  float*    rmax = wh1 + ROWS;                   // [ROWS]
  float*    rinv = rmax + ROWS;                  // [ROWS]
  _Float16* px   = (_Float16*)(rinv + ROWS);     // [ROWS][PXS]   2.5 KB (16B aligned)
  _Float16* xbT  = px + ROWS * PXS;              // [GH][XTS]     5 KB  (16B aligned)

  const int b    = blockIdx.y;
  const int p0   = blockIdx.x * ROWS;
  const int tid  = threadIdx.x;
  const int wave = tid >> 5;
  const int lane = tid & 31;

  // ---- TDM: DMA the contiguous 32x1024 f32 A tile + wh2 row into LDS
  if (wave == 0) {
    tdm_load_1d((unsigned)(uintptr_t)(void*)e,
                Adj + ((size_t)b * GN + p0) * GN, ROWS * GN);
    tdm_load_1d((unsigned)(uintptr_t)(void*)wh2,
                wh2g + (size_t)b * GN, GN);
    __builtin_amdgcn_s_wait_tensorcnt(0);
  }
  if (tid < ROWS) wh1[tid] = wh1g[(size_t)b * GN + p0 + tid];
  __syncthreads();

  // ---- Pass A: transform raw adjacency in place, row max / sum(exp)
  for (int rr = 0; rr < ROWS / 4; ++rr) {        // 8 rows per wave
    int row = wave * (ROWS / 4) + rr;
    float w1 = wh1[row];
    float lmax = -3.0e38f;
    for (int q = lane; q < GN; q += 32) {
      float a = e[row * GN + q];                 // raw A (TDM-staged)
      float s = w1 + wh2[q];
      float ev = (a > 0.f) ? ((s >= 0.f) ? s : 0.2f * s) : -1.0e12f;
      e[row * GN + q] = ev;
      lmax = fmaxf(lmax, ev);
    }
    #pragma unroll
    for (int off = 16; off; off >>= 1)
      lmax = fmaxf(lmax, __shfl_xor(lmax, off, 32));
    float lsum = 0.f;
    for (int q = lane; q < GN; q += 32)
      lsum += __expf(e[row * GN + q] - lmax);
    #pragma unroll
    for (int off = 16; off; off >>= 1)
      lsum += __shfl_xor(lsum, off, 32);
    if (lane == 0) { rmax[row] = lmax; rinv[row] = 1.f / lsum; }
  }
  __syncthreads();

  // ---- Pass B: chunked attn @ inp via WMMA
  const int rt    = wave >> 1;          // 16-row tile: 0 or 1
  const int hbase = (wave & 1) * 32;    // this wave's 2 output n-tiles
  const int mrow  = lane & 15;          // A-frag row / B-frag col / D col
  const int hsel  = lane >> 4;          // lane-half selector

  v8f acc0, acc1;
  #pragma unroll
  for (int i = 0; i < 8; ++i) { acc0[i] = 0.f; acc1[i] = 0.f; }

  for (int k0 = 0; k0 < GN; k0 += CHUNK) {
    __syncthreads();  // px/xbT reuse from previous chunk
    // probabilities tile (each exp computed once, shared by waves)
    for (int i = tid; i < ROWS * CHUNK; i += THREADS) {
      int r = i >> 5, c = i & 31;
      float ev = e[r * GN + k0 + c];
      px[r * PXS + c] = (_Float16)(__expf(ev - rmax[r]) * rinv[r]);
    }
    // inp chunk (L2-resident) -> transposed f16 LDS: xbT[col][k]
    const float* X = inp + ((size_t)b * GN + k0) * GH;
    for (int i = tid; i < CHUNK * GH; i += THREADS) {
      __builtin_prefetch(X + i + CHUNK * GH, 0, 0);  // global_prefetch_b8
      int k = i >> 6, n = i & 63;                    // coalesced global read
      xbT[n * XTS + k] = (_Float16)X[i];
    }
    __syncthreads();

    // A-frag (16x32): two aligned b128 LDS loads (K run hsel*16..+15)
    const _Float16* prow = px + (rt * 16 + mrow) * PXS;
    v8h alo = *(const v8h*)(prow + hsel * 16);
    v8h ahi = *(const v8h*)(prow + hsel * 16 + 8);
    v16h af = __builtin_shufflevector(alo, ahi, 0, 1, 2, 3, 4, 5, 6, 7,
                                      8, 9, 10, 11, 12, 13, 14, 15);
    // B-frags (32x16): contiguous K run per lane -> two aligned b128 loads each
    const _Float16* c0 = xbT + (hbase + mrow) * XTS + hsel * 16;
    const _Float16* c1 = xbT + (hbase + 16 + mrow) * XTS + hsel * 16;
    v8h b0l = *(const v8h*)(c0);
    v8h b0h = *(const v8h*)(c0 + 8);
    v8h b1l = *(const v8h*)(c1);
    v8h b1h = *(const v8h*)(c1 + 8);
    v16h bf0 = __builtin_shufflevector(b0l, b0h, 0, 1, 2, 3, 4, 5, 6, 7,
                                       8, 9, 10, 11, 12, 13, 14, 15);
    v16h bf1 = __builtin_shufflevector(b1l, b1h, 0, 1, 2, 3, 4, 5, 6, 7,
                                       8, 9, 10, 11, 12, 13, 14, 15);

    acc0 = __builtin_amdgcn_wmma_f32_16x16x32_f16(
        false, af, false, bf0, (short)0, acc0, false, false);
    acc1 = __builtin_amdgcn_wmma_f32_16x16x32_f16(
        false, af, false, bf1, (short)0, acc1, false, false);
  }

  // ---- Epilogue: elu, direct stores (C/D layout: VGPR r -> rows r, r+8)
  #pragma unroll
  for (int r = 0; r < 8; ++r) {
    int grow = p0 + rt * 16 + hsel * 8 + r;
    size_t base = ((size_t)b * GN + grow) * GH;
    float x0 = acc0[r];
    float x1 = acc1[r];
    out[base + hbase + mrow]      = (x0 > 0.f) ? x0 : (__expf(x0) - 1.f);
    out[base + hbase + 16 + mrow] = (x1 > 0.f) ? x1 : (__expf(x1) - 1.f);
  }
}

// ---------------------------------------------------------------------------
extern "C" void kernel_launch(void* const* d_in, const int* in_sizes, int n_in,
                              void* d_out, int out_size, void* d_ws, size_t ws_size,
                              hipStream_t stream) {
  (void)in_sizes; (void)n_in; (void)out_size; (void)ws_size;
  const float* inp      = (const float*)d_in[0];
  const float* inp_none = (const float*)d_in[1];
  const float* Adj      = (const float*)d_in[2];
  const int*   l        = (const int*)d_in[3];
  const float* a_i      = (const float*)d_in[4];
  const float* a_c      = (const float*)d_in[5];
  float* out = (float*)d_out;

  float* wh1 = (float*)d_ws;           // [B][N]
  float* wh2 = wh1 + GB * GN;          // [B][N]  (512 KB total workspace)

  gat_scores<<<dim3((GB * GN) / THREADS), THREADS, 0, stream>>>(
      inp, inp_none, l, a_i, a_c, wh1, wh2);

  size_t smem = (size_t)(ROWS * GN + GN + 3 * ROWS) * sizeof(float)
              + (size_t)(ROWS * PXS + GH * XTS) * 2;   // ~140 KB
  (void)hipFuncSetAttribute((const void*)gat_main,
                            hipFuncAttributeMaxDynamicSharedMemorySize,
                            (int)smem);
  gat_main<<<dim3(GN / ROWS, GB), THREADS, smem, stream>>>(
      inp, Adj, wh1, wh2, out);
}
